// Memory_63642825392438
// MI455X (gfx1250) — matrix operand order repaired and checked by reading, not compiled, not run on hardware
//
#include <hip/hip_runtime.h>

// ---------------- problem constants ----------------
#define B_    4
#define DE    128
#define DO    512
#define HW    1560
#define HW16  1568          // HW padded to multiple of 16 (98 tiles)
#define THW   6240          // = 195*32 = 390*16
#define NPIX  (B_ * HW16)   // 6272 flat padded pixels
#define SCALE 0.08838834764831843f   // 1/sqrt(128)

typedef __attribute__((ext_vector_type(16))) __bf16 v16bf;
typedef __attribute__((ext_vector_type(8)))  float  v8f;

union ABu { v16bf v; uint4 q[2]; unsigned short us[16]; };
union F8  { float f[8]; float4 q[2]; };

__device__ __forceinline__ unsigned short f2bf(float f) {
  unsigned int u = __float_as_uint(f);
  u += 0x7FFFu + ((u >> 16) & 1u);       // round-to-nearest-even
  return (unsigned short)(u >> 16);
}

__device__ __forceinline__ v8f zero8() {
  v8f z;
#pragma unroll
  for (int i = 0; i < 8; ++i) z[i] = 0.f;
  return z;
}

__device__ __forceinline__ v8f wmma_bf16(v16bf a, v16bf b, v8f c) {
  return __builtin_amdgcn_wmma_f32_16x16x32_bf16(false, a, false, b, (short)0, c,
                                                 false, false);
}

// A operand 16x32 bf16, source row-major [rows][ld]:
// lane<16: row=lane, K = 0..15; lane>=16: row=lane-16, K=16..31
__device__ __forceinline__ v16bf load_a16(const unsigned short* __restrict__ base,
                                          int ld, int row0, int k0, int lane) {
  int r = lane & 15, h = lane >> 4;
  const unsigned short* p = base + (size_t)(row0 + r) * ld + k0 + h * 16;
  ABu t;
  t.q[0] = *(const uint4*)p;
  t.q[1] = *(const uint4*)(p + 8);
  return t.v;
}

// B operand 32x16 bf16, source "col-major" X[col][k] with row length ld:
// lane<16: col=lane, K=0..15; lane>=16: col=lane-16, K=16..31
__device__ __forceinline__ v16bf load_b16(const unsigned short* __restrict__ base,
                                          int ld, int col0, int k0, int lane) {
  int c = lane & 15, h = lane >> 4;
  const unsigned short* p = base + (size_t)(col0 + c) * ld + k0 + h * 16;
  ABu t;
  t.q[0] = *(const uint4*)p;
  t.q[1] = *(const uint4*)(p + 8);
  return t.v;
}

// B operand from an LDS panel laid out [16 pixels][DO channels] bf16
__device__ __forceinline__ v16bf load_b16_lds(const unsigned short* base,
                                              int k0, int lane) {
  int c = lane & 15, h = lane >> 4;
  const unsigned short* p = base + c * DO + k0 + h * 16;
  ABu t;
  t.q[0] = *(const uint4*)p;
  t.q[1] = *(const uint4*)(p + 8);
  return t.v;
}

__device__ __forceinline__ float fsigmoid(float x) {
  x = fminf(fmaxf(x, -30.f), 30.f);
  return 1.0f / (1.0f + __expf(-x));
}
__device__ __forceinline__ float ftanh_(float x) {
  x = fminf(fmaxf(x, -15.f), 15.f);
  float e = __expf(-2.f * x);
  return (1.f - e) / (1.f + e);
}

// ---------------- prep / layout kernels ----------------

// A_kd[b][k][d] = m_in[b][d][k]  (bf16, key rows row-major over De)
__global__ void prep_m_kernel(const float* __restrict__ m_in,
                              unsigned short* __restrict__ Akd, int n) {
  int i = blockIdx.x * 256 + threadIdx.x;
  if (i >= n) return;
  int d = i % DE, k = (i / DE) % THW, b = i / (DE * THW);
  Akd[i] = f2bf(m_in[((size_t)b * DE + d) * THW + k]);
}

// qiT[b][q][d] = q_in[b][d][q] (bf16, zero-padded q in [HW,HW16))
__global__ void prep_q_kernel(const float* __restrict__ q_in,
                              unsigned short* __restrict__ qiT, int n) {
  int i = blockIdx.x * 256 + threadIdx.x;
  if (i >= n) return;
  int d = i % DE, q = (i / DE) % HW16, b = i / (DE * HW16);
  float v = (q < HW) ? q_in[((size_t)b * DE + d) * HW + q] : 0.f;
  qiT[i] = f2bf(v);
}

// mo_bf: straight f32->bf16 convert, same [B][Do][THW] layout
__global__ void prep_mo_kernel(const float* __restrict__ m_out,
                               unsigned short* __restrict__ mo, int n) {
  int i = blockIdx.x * 256 + threadIdx.x;
  if (i >= n) return;
  mo[i] = f2bf(m_out[i]);
}

// three gate weights [Do][2Do] -> bf16 packed consecutively
__global__ void prep_w_kernel(const float* __restrict__ wr,
                              const float* __restrict__ wu,
                              const float* __restrict__ wc,
                              unsigned short* __restrict__ wbf, int n) {
  int i = blockIdx.x * 256 + threadIdx.x;
  if (i >= n) return;
  int j = i % (DO * 2 * DO), w = i / (DO * 2 * DO);
  const float* s = (w == 0) ? wr : (w == 1) ? wu : wc;
  wbf[i] = f2bf(s[j]);
}

// initial state: q_T[pix][o] (f32) and q_bfT[pix][o] (bf16) from q_out [B][Do][HW]
__global__ void prep_qout_kernel(const float* __restrict__ q_out,
                                 float* __restrict__ qT,
                                 unsigned short* __restrict__ qBF, int n) {
  int i = blockIdx.x * 256 + threadIdx.x;
  if (i >= n) return;
  int o = i % DO, q = (i / DO) % HW16, b = i / (DO * HW16);
  float v = (q < HW) ? q_out[((size_t)b * DO + o) * HW + q] : 0.f;
  qT[i] = v;
  qBF[i] = f2bf(v);
}

// out[b][ch][hw] = ch<Do ? q_T[b*HW16+hw][ch] : q_out[b][ch-Do][hw]
__global__ void final_kernel(const float* __restrict__ qT,
                             const float* __restrict__ q_out,
                             float* __restrict__ out, int n) {
  int i = blockIdx.x * 256 + threadIdx.x;
  if (i >= n) return;
  int hw = i % HW, ch = (i / HW) % (2 * DO), b = i / (2 * DO * HW);
  float v;
  if (ch < DO) v = qT[((size_t)b * HW16 + hw) * DO + ch];
  else         v = q_out[((size_t)b * DO + (ch - DO)) * HW + hw];
  out[i] = v;
}

// ---------------- attention pass 1: softmax stats (streamed scores) ----------------
// one block per (b, q-tile); 8 waves split the 390 key tiles, online (max,sum)
__global__ void __launch_bounds__(256) stats_kernel(
    const unsigned short* __restrict__ Akd, const unsigned short* __restrict__ qiT,
    float* __restrict__ rowmax, float* __restrict__ rowsum) {
  int b = blockIdx.x / (HW16 / 16);
  int q0 = (blockIdx.x % (HW16 / 16)) * 16;
  int wave = threadIdx.x >> 5, lane = threadIdx.x & 31;
  const unsigned short* A = Akd + (size_t)b * THW * DE;
  const unsigned short* Q = qiT + (size_t)b * HW16 * DE;
  v16bf btile[4];
#pragma unroll
  for (int d = 0; d < 4; ++d) btile[d] = load_b16(Q, DE, q0, d * 32, lane);
  float m = -3.0e38f, l = 0.f;
  for (int kt = wave; kt < THW / 16; kt += 8) {
    v8f acc = zero8();
#pragma unroll
    for (int d = 0; d < 4; ++d)
      acc = wmma_bf16(load_a16(A, DE, kt * 16, d * 32, lane), btile[d], acc);
    float tm = -3.0e38f;
#pragma unroll
    for (int v = 0; v < 8; ++v) tm = fmaxf(tm, acc[v] * SCALE);
    float mn = fmaxf(m, tm);
    l *= __expf(m - mn);
#pragma unroll
    for (int v = 0; v < 8; ++v) l += __expf(acc[v] * SCALE - mn);
    m = mn;
  }
  // combine lane pair (same column lives in lane and lane+16)
  float om = __shfl_xor(m, 16), ol = __shfl_xor(l, 16);
  float mn = fmaxf(m, om);
  l = l * __expf(m - mn) + ol * __expf(om - mn);
  m = mn;
  __shared__ float sm[8][16], sl[8][16];
  if (lane < 16) { sm[wave][lane] = m; sl[wave][lane] = l; }
  __syncthreads();
  if (threadIdx.x < 16) {
    float M = -3.0e38f, L = 0.f;
    for (int w = 0; w < 8; ++w) {
      float mw = sm[w][threadIdx.x], lw = sl[w][threadIdx.x];
      float mm = fmaxf(M, mw);
      L = L * __expf(M - mm) + lw * __expf(mw - mm);
      M = mm;
    }
    rowmax[b * HW16 + q0 + threadIdx.x] = M;
    rowsum[b * HW16 + q0 + threadIdx.x] = L;
  }
}

// ---------------- attention pass 2: mem = m_out * softmax(s) ----------------
// one block per (b, q-tile). Wave w produces p super-tile (32k x 16q) for key
// super-tile r*8+w into LDS; after barrier every wave consumes all 8 slots for
// its private 64 output rows (o = wave*64..+63). No score redundancy.
__global__ void __launch_bounds__(256) memv_kernel(
    const unsigned short* __restrict__ Akd, const unsigned short* __restrict__ qiT,
    const unsigned short* __restrict__ MO,
    const float* __restrict__ rowmax, const float* __restrict__ rowsum,
    float* __restrict__ memT, unsigned short* __restrict__ memBF) {
  __shared__ unsigned short pbuf[8][16][32];   // [slot][q][k], q-major
  int b = blockIdx.x / (HW16 / 16);
  int q0 = (blockIdx.x % (HW16 / 16)) * 16;
  int wave = threadIdx.x >> 5, lane = threadIdx.x & 31;
  int cc = lane & 15, h = lane >> 4;
  const unsigned short* A = Akd + (size_t)b * THW * DE;
  const unsigned short* Q = qiT + (size_t)b * HW16 * DE;
  const unsigned short* V = MO + (size_t)b * DO * THW;
  float mq = rowmax[b * HW16 + q0 + cc];
  float invl = 1.0f / rowsum[b * HW16 + q0 + cc];
  v16bf btile[4];
#pragma unroll
  for (int d = 0; d < 4; ++d) btile[d] = load_b16(Q, DE, q0, d * 32, lane);
  v8f accs[4];
#pragma unroll
  for (int ot = 0; ot < 4; ++ot) accs[ot] = zero8();
  int o0 = wave * 64;
  const int NST = THW / 32;   // 195 key super-tiles of 32
  for (int r = 0; r < (NST + 7) / 8; ++r) {
    int st = r * 8 + wave;
    if (st < NST) {
      v8f s0 = zero8(), s1 = zero8();
#pragma unroll
      for (int d = 0; d < 4; ++d) {
        s0 = wmma_bf16(load_a16(A, DE, st * 32,      d * 32, lane), btile[d], s0);
        s1 = wmma_bf16(load_a16(A, DE, st * 32 + 16, d * 32, lane), btile[d], s1);
      }
      unsigned short* dst = &pbuf[wave][cc][0];
      int rb = h * 8;   // C layout: lanes 0-15 rows 0-7, lanes 16-31 rows 8-15
#pragma unroll
      for (int v = 0; v < 8; ++v) {
        dst[rb + v]      = f2bf(__expf(s0[v] * SCALE - mq));
        dst[16 + rb + v] = f2bf(__expf(s1[v] * SCALE - mq));
      }
    }
    __syncthreads();
    int nst = NST - r * 8; if (nst > 8) nst = 8;
    for (int s = 0; s < nst; ++s) {
      const unsigned short* pp = &pbuf[s][cc][h * 16];
      ABu t;
      t.q[0] = *(const uint4*)pp;
      t.q[1] = *(const uint4*)(pp + 8);
      int kst = (r * 8 + s) * 32;
#pragma unroll
      for (int ot = 0; ot < 4; ++ot)
        accs[ot] = wmma_bf16(load_a16(V, THW, o0 + ot * 16, kst, lane), t.v, accs[ot]);
    }
    __syncthreads();
  }
#pragma unroll
  for (int ot = 0; ot < 4; ++ot) {
    size_t idx = ((size_t)(b * HW16 + q0 + cc)) * DO + o0 + ot * 16 + h * 8;
    F8 o; ABu t;
#pragma unroll
    for (int v = 0; v < 8; ++v) {
      float val = accs[ot][v] * invl;
      o.f[v] = val;
      t.us[v] = f2bf(val);
    }
    *(float4*)(memT + idx)     = o.q[0];
    *(float4*)(memT + idx + 4) = o.q[1];
    *(uint4*)(memBF + idx)     = t.q[0];
  }
}

// ---------------- GRU gate GEMM with fused epilogues ----------------
// All 8 waves of a block share one 16-pixel tile; the activation panel(s)
// ([16 pix][512 ch] bf16, 16KB each) are staged cooperatively into LDS once,
// so B-operand reads are ds_load_b128 instead of 8x-redundant global loads.
// mode 0: cr = Wr_h*mem + br          (W cols 512.., X1=mem_bf) -> outF
// mode 1: cu = Wu_h*mem + bu          -> outF
// mode 2: r = sig(Wr_x*q + cr); rh_bf = bf16(r*mem)  -> outBF
// mode 3: u = sig(Wu_x*q + cu)        -> outF
// mode 4: c = tanh(Wc_x*q + Wc_h*rh + bc); q' = mem + u*(c-mem) -> outF + outBF
__global__ void __launch_bounds__(256) gate_kernel(
    int mode,
    const unsigned short* __restrict__ W,    // [Do][2Do] bf16 row-major
    const unsigned short* __restrict__ X1,   // [NPIX][Do] bf16
    const unsigned short* __restrict__ X2,   // [NPIX][Do] bf16 (mode 4)
    const float* __restrict__ bias,
    const float* __restrict__ addT,
    const float* __restrict__ memT,
    const float* __restrict__ uT,
    float* __restrict__ outF,
    unsigned short* __restrict__ outBF) {
  __shared__ unsigned short xs1[16 * DO];   // 16KB
  __shared__ unsigned short xs2[16 * DO];   // 16KB (written only in mode 4)
  int wave = threadIdx.x >> 5, lane = threadIdx.x & 31;
  int gw = blockIdx.x * 8 + wave;
  int o0 = (gw & 31) * 16;          // 32 o-tiles (waves of a block: consecutive)
  int p0 = (gw >> 5) * 16;          // 392 pixel tiles (uniform per block)
  int c0 = (mode <= 1) ? DO : 0;

  // prefetch this wave's weight panel region (L2 warm-up): global_prefetch_b8
  __builtin_prefetch(W + (size_t)(o0 + (lane & 15)) * (2 * DO) + c0 + (lane >> 4) * 256,
                     0, 0);

  // cooperative stage of X panel(s): thread t loads 64B of pixel t/16
  {
    int pix = threadIdx.x >> 4;
    int ch  = (threadIdx.x & 15) * 32;
    const uint4* src = (const uint4*)(X1 + (size_t)(p0 + pix) * DO + ch);
    uint4* dst = (uint4*)(xs1 + pix * DO + ch);
#pragma unroll
    for (int j = 0; j < 4; ++j) dst[j] = src[j];
    if (mode == 4) {
      const uint4* s2 = (const uint4*)(X2 + (size_t)(p0 + pix) * DO + ch);
      uint4* d2 = (uint4*)(xs2 + pix * DO + ch);
#pragma unroll
      for (int j = 0; j < 4; ++j) d2[j] = s2[j];
    }
  }
  __syncthreads();

  v8f acc = zero8();
  for (int kt = 0; kt < DO / 32; ++kt)
    acc = wmma_bf16(load_a16(W, 2 * DO, o0, c0 + kt * 32, lane),
                    load_b16_lds(xs1, kt * 32, lane), acc);
  if (mode == 4) {
    for (int kt = 0; kt < DO / 32; ++kt)
      acc = wmma_bf16(load_a16(W, 2 * DO, o0, DO + kt * 32, lane),
                      load_b16_lds(xs2, kt * 32, lane), acc);
  }
  int cc = lane & 15, h = lane >> 4;
  size_t idx = ((size_t)(p0 + cc)) * DO + o0 + h * 8;
  if (mode <= 1) {
    const float* bp = bias + o0 + h * 8;
    F8 o;
#pragma unroll
    for (int v = 0; v < 8; ++v) o.f[v] = acc[v] + bp[v];
    *(float4*)(outF + idx)     = o.q[0];
    *(float4*)(outF + idx + 4) = o.q[1];
  } else if (mode == 2) {
    ABu t;
#pragma unroll
    for (int v = 0; v < 8; ++v) {
      float r = fsigmoid(acc[v] + addT[idx + v]);
      t.us[v] = f2bf(r * memT[idx + v]);
    }
    *(uint4*)(outBF + idx) = t.q[0];
  } else if (mode == 3) {
    F8 o;
#pragma unroll
    for (int v = 0; v < 8; ++v) o.f[v] = fsigmoid(acc[v] + addT[idx + v]);
    *(float4*)(outF + idx)     = o.q[0];
    *(float4*)(outF + idx + 4) = o.q[1];
  } else {
    const float* bp = bias + o0 + h * 8;
    F8 o; ABu t;
#pragma unroll
    for (int v = 0; v < 8; ++v) {
      float cv = ftanh_(acc[v] + bp[v]);
      float u = uT[idx + v], mm = memT[idx + v];
      float qn = mm + u * (cv - mm);
      o.f[v] = qn;
      t.us[v] = f2bf(qn);
    }
    *(float4*)(outF + idx)     = o.q[0];
    *(float4*)(outF + idx + 4) = o.q[1];
    *(uint4*)(outBF + idx)     = t.q[0];
  }
}

// ---------------- host launch ----------------
extern "C" void kernel_launch(void* const* d_in, const int* in_sizes, int n_in,
                              void* d_out, int out_size, void* d_ws, size_t ws_size,
                              hipStream_t stream) {
  (void)in_sizes; (void)n_in; (void)out_size; (void)ws_size;
  const float* m_in  = (const float*)d_in[0];
  const float* m_out = (const float*)d_in[1];
  const float* q_in  = (const float*)d_in[2];
  const float* q_out = (const float*)d_in[3];
  const float* wr    = (const float*)d_in[4];
  const float* br    = (const float*)d_in[5];
  const float* wu    = (const float*)d_in[6];
  const float* bu    = (const float*)d_in[7];
  const float* wc    = (const float*)d_in[8];
  const float* bc    = (const float*)d_in[9];

  char* ws = (char*)d_ws;
  size_t off = 0;
  auto alloc = [&](size_t bytes) {
    size_t o = off;
    off += (bytes + 255) & ~(size_t)255;
    return o;
  };
  unsigned short* Akd  = (unsigned short*)(ws + alloc((size_t)B_ * THW * DE * 2));
  unsigned short* QiT  = (unsigned short*)(ws + alloc((size_t)B_ * HW16 * DE * 2));
  unsigned short* Mo   = (unsigned short*)(ws + alloc((size_t)B_ * DO * THW * 2));
  unsigned short* Wbf  = (unsigned short*)(ws + alloc((size_t)3 * DO * 2 * DO * 2));
  float*          Max  = (float*)(ws + alloc((size_t)B_ * HW16 * 4));
  float*          Sum  = (float*)(ws + alloc((size_t)B_ * HW16 * 4));
  float*          MemF = (float*)(ws + alloc((size_t)NPIX * DO * 4));
  unsigned short* MemB = (unsigned short*)(ws + alloc((size_t)NPIX * DO * 2));
  float*          Cr   = (float*)(ws + alloc((size_t)NPIX * DO * 4));
  float*          Cu   = (float*)(ws + alloc((size_t)NPIX * DO * 4));
  float*          Uf   = (float*)(ws + alloc((size_t)NPIX * DO * 4));
  float*          Qf   = (float*)(ws + alloc((size_t)NPIX * DO * 4));
  unsigned short* Qba  = (unsigned short*)(ws + alloc((size_t)NPIX * DO * 2));
  unsigned short* Qbb  = (unsigned short*)(ws + alloc((size_t)NPIX * DO * 2));
  unsigned short* Rh   = (unsigned short*)(ws + alloc((size_t)NPIX * DO * 2));

  int n;
  n = B_ * THW * DE;
  prep_m_kernel<<<(n + 255) / 256, 256, 0, stream>>>(m_in, Akd, n);
  n = B_ * HW16 * DE;
  prep_q_kernel<<<(n + 255) / 256, 256, 0, stream>>>(q_in, QiT, n);
  n = B_ * DO * THW;
  prep_mo_kernel<<<(n + 255) / 256, 256, 0, stream>>>(m_out, Mo, n);
  n = 3 * DO * 2 * DO;
  prep_w_kernel<<<(n + 255) / 256, 256, 0, stream>>>(wr, wu, wc, Wbf, n);
  n = B_ * HW16 * DO;
  prep_qout_kernel<<<(n + 255) / 256, 256, 0, stream>>>(q_out, Qf, Qba, n);

  stats_kernel<<<B_ * (HW16 / 16), 256, 0, stream>>>(Akd, QiT, Max, Sum);
  memv_kernel<<<B_ * (HW16 / 16), 256, 0, stream>>>(Akd, QiT, Mo, Max, Sum, MemF, MemB);

  const int GB = (NPIX / 16) * (DO / 16) / 8;   // 1568 blocks of 8 wave-tiles
  const unsigned short* Wr = Wbf;
  const unsigned short* Wu = Wbf + (size_t)DO * 2 * DO;
  const unsigned short* Wc = Wbf + (size_t)2 * DO * 2 * DO;
  // iteration-invariant halves
  gate_kernel<<<GB, 256, 0, stream>>>(0, Wr, MemB, nullptr, br, nullptr, nullptr,
                                      nullptr, Cr, nullptr);
  gate_kernel<<<GB, 256, 0, stream>>>(1, Wu, MemB, nullptr, bu, nullptr, nullptr,
                                      nullptr, Cu, nullptr);
  for (int it = 0; it < 5; ++it) {
    unsigned short* Qin  = (it & 1) ? Qbb : Qba;
    unsigned short* Qout = (it & 1) ? Qba : Qbb;
    gate_kernel<<<GB, 256, 0, stream>>>(2, Wr, Qin, nullptr, nullptr, Cr, MemF,
                                        nullptr, nullptr, Rh);
    gate_kernel<<<GB, 256, 0, stream>>>(3, Wu, Qin, nullptr, nullptr, Cu, nullptr,
                                        nullptr, Uf, nullptr);
    gate_kernel<<<GB, 256, 0, stream>>>(4, Wc, Qin, Rh, bc, nullptr, MemF, Uf,
                                        Qf, Qout);
  }
  n = B_ * 2 * DO * HW;
  final_kernel<<<(n + 255) / 256, 256, 0, stream>>>(Qf, q_out, (float*)d_out, n);
}
